// DownsampleConv2dGT_81389630259536
// MI455X (gfx1250) — matrix operand order repaired and checked by reading, not compiled, not run on hardware
//
#include <hip/hip_runtime.h>

#define B_   16
#define T_   2048
#define D_   80
#define HID_ 64
#define OUT_ 512
#define NB_  64
#define DOWN_ 4.0

typedef __attribute__((ext_vector_type(16))) _Float16     v16h;
typedef __attribute__((ext_vector_type(8)))  float        v8f;
typedef __attribute__((ext_vector_type(4)))  unsigned int u32x4;

union Frag16 { v16h h; u32x4 q[2]; };

// 16 contiguous halfs (B operand): two 16B loads
__device__ __forceinline__ v16h load_frag16(const _Float16* p) {
  Frag16 f;
  f.q[0] = *reinterpret_cast<const u32x4*>(p);
  f.q[1] = *reinterpret_cast<const u32x4*>(p + 8);
  return f.h;
}
// A operand: two 8-half chunks 8 elements apart in K (16 contiguous k)
__device__ __forceinline__ v16h load_frag8x2(const _Float16* p0, const _Float16* p1) {
  Frag16 f;
  f.q[0] = *reinterpret_cast<const u32x4*>(p0);
  f.q[1] = *reinterpret_cast<const u32x4*>(p1);
  return f.h;
}
// A fragment from LDS (ds_load_b128 x2)
__device__ __forceinline__ v16h lds_frag8x2(const _Float16* p0, const _Float16* p1) {
  Frag16 f;
  f.q[0] = *reinterpret_cast<const u32x4*>(p0);
  f.q[1] = *reinterpret_cast<const u32x4*>(p1);
  return f.h;
}
__device__ __forceinline__ v16h zero_frag() {
  Frag16 f;
  f.q[0] = (u32x4){0u,0u,0u,0u};
  f.q[1] = (u32x4){0u,0u,0u,0u};
  return f.h;
}

// ---------------- weight prep ----------------
__global__ void prep_wpT(const float* __restrict__ wp, _Float16* __restrict__ wpT) {
  int idx = blockIdx.x * blockDim.x + threadIdx.x;
  const int K = HID_ * D_; // 5120
  if (idx >= K * OUT_) return;
  const int n = idx % OUT_;
  const int k = idx / OUT_;
  wpT[(size_t)n * K + k] = (_Float16)wp[idx]; // wp is [k][n]
}

__global__ void prep_w2p(const float* __restrict__ w2, _Float16* __restrict__ w2p) {
  int idx = blockIdx.x * blockDim.x + threadIdx.x;
  if (idx >= 9 * HID_ * HID_) return;
  const int ci  = idx % HID_;
  const int co  = (idx / HID_) % HID_;
  const int tap = idx / (HID_ * HID_);
  // w2 layout (CO, CI, 3, 3) -> w2p [tap][co][ci]
  w2p[((size_t)tap * HID_ + co) * HID_ + ci] =
      (_Float16)w2[((size_t)co * HID_ + ci) * 9 + tap];
}

// ---------------- segment construction (replicates host python) ----------------
__global__ void seg_build(const int* __restrict__ x_len,
                          const int* __restrict__ bnd,
                          int* __restrict__ seg_s, int* __restrict__ seg_e,
                          int* __restrict__ out_len_dst, int S, int write_len) {
  const int b = threadIdx.x;
  if (b >= B_) return;
  const int L = x_len[b];
  const double rate = (DOWN_ > 1.0) ? ((double)L / (double)NB_ / DOWN_) : 1.0;
  const int lastb = bnd[b * NB_ + NB_ - 1];
  const int nbd = NB_ + ((lastb < L) ? 1 : 0);
  int cnt = 0, prev = 0;
  for (int i = 0; i < nbd; ++i) {
    int t = (i < NB_) ? bnd[b * NB_ + i] : L;
    if (t > L) t = L;
    if (prev == t) continue;
    int delta = (int)rint((double)(t - prev) / rate); // banker's rounding like python
    if (delta <= 0) {
      if (cnt < S) { seg_s[b * S + cnt] = prev; seg_e[b * S + cnt] = t; }
      ++cnt;
    } else {
      int _t = prev;
      while (_t < t) {
        int e = _t + delta; if (e > t) e = t;
        if (cnt < S) { seg_s[b * S + cnt] = _t; seg_e[b * S + cnt] = e; }
        ++cnt; _t += delta;
      }
    }
    prev = t;
  }
  for (int ss = cnt; ss < S; ++ss) { seg_s[b * S + ss] = 0; seg_e[b * S + ss] = 0; }
  if (write_len && out_len_dst) out_len_dst[b] = cnt;
}

// ---------------- segment-mean pooling ----------------
__global__ void pool_kernel(const float* __restrict__ x,
                            const int* __restrict__ seg_s,
                            const int* __restrict__ seg_e,
                            float* __restrict__ pooled, int S) {
  int idx = blockIdx.x * blockDim.x + threadIdx.x;
  const int total = B_ * S * D_;
  if (idx >= total) return;
  const int d = idx % D_;
  const int s = (idx / D_) % S;
  const int b = idx / (D_ * S);
  const int st = seg_s[b * S + s];
  const int en = seg_e[b * S + s];
  float acc = 0.f;
  if (en > st) {
    for (int t = st; t < en; ++t) acc += x[((size_t)b * T_ + t) * D_ + d];
    acc *= 1.f / (float)(en - st);
  }
  pooled[idx] = acc;
}

// ---------------- conv1: 1 -> 64 channels, 3x3, relu; channels-last f16 out ----------------
__global__ void conv1_kernel(const float* __restrict__ pooled,
                             const float* __restrict__ w1,
                             const float* __restrict__ b1,
                             _Float16* __restrict__ h1, int S) {
  __shared__ float w1s[HID_ * 9];
  for (int i = threadIdx.x; i < HID_ * 9; i += blockDim.x) w1s[i] = w1[i];
  __syncthreads();
  int idx = blockIdx.x * blockDim.x + threadIdx.x;
  const int total = B_ * S * D_;
  if (idx >= total) return;
  const int d = idx % D_;
  const int s = (idx / D_) % S;
  const int b = idx / (D_ * S);

  float in[9];
#pragma unroll
  for (int i = 0; i < 3; ++i)
#pragma unroll
    for (int j = 0; j < 3; ++j) {
      const int sp = s + i - 1, dp = d + j - 1;
      in[i * 3 + j] = (sp >= 0 && sp < S && dp >= 0 && dp < D_)
                          ? pooled[((size_t)b * S + sp) * D_ + dp] : 0.f;
    }
  _Float16* dst = h1 + (size_t)idx * HID_;
  for (int c = 0; c < HID_; ++c) {
    float acc = b1[c];
#pragma unroll
    for (int t = 0; t < 9; ++t) acc = fmaf(in[t], w1s[c * 9 + t], acc);
    acc = acc > 0.f ? acc : 0.f;
    dst[c] = (_Float16)acc;
  }
}

// -------- conv2 as WMMA GEMM: per wave 16 positions x 64 out-channels, A reused 4x --------
__global__ void conv2_wmma(const _Float16* __restrict__ h1,
                           const _Float16* __restrict__ w2p,
                           const float* __restrict__ b2,
                           _Float16* __restrict__ h2, int S) {
  const int lane = threadIdx.x;
  const int m  = lane & 15;   // row-in-tile / column selector
  const int hi = lane >> 4;   // 0/1: K-chunk half per 16-bit A/B layout
  int mt = blockIdx.x;
  const int dq = mt % 5; mt /= 5;     // 80 = 5 tiles of 16 along D
  const int s  = mt % S;
  const int b  = mt / S;

  v8f zero8 = {0.f,0.f,0.f,0.f,0.f,0.f,0.f,0.f};
  v8f acc[4];
#pragma unroll
  for (int j = 0; j < 4; ++j) acc[j] = zero8;

#pragma unroll
  for (int ds = -1; ds <= 1; ++ds) {
    const int sp = s + ds;
    if (sp < 0 || sp >= S) continue;
#pragma unroll
    for (int dd = -1; dd <= 1; ++dd) {
      const int tap = (ds + 1) * 3 + (dd + 1);
      const int d = dq * 16 + m + dd;
      const bool ok = (d >= 0) && (d < D_);
      const _Float16* abase =
          h1 + ((size_t)((b * S + sp) * D_ + (ok ? d : 0)) * HID_);
      // B: w2p[tap][co][ci]; lane holds column co = j*16 + m
      const _Float16* wb = w2p + ((size_t)tap * HID_ + m) * HID_;
#pragma unroll
      for (int kc = 0; kc < 2; ++kc) {
        const int ka = kc * 32 + hi * 16;
        const int kb = kc * 32 + hi * 16;
        v16h A  = ok ? load_frag8x2(abase + ka, abase + ka + 8) : zero_frag();
        v16h B0 = load_frag16(wb + 0 * 16 * HID_ + kb);
        v16h B1 = load_frag16(wb + 1 * 16 * HID_ + kb);
        v16h B2 = load_frag16(wb + 2 * 16 * HID_ + kb);
        v16h B3 = load_frag16(wb + 3 * 16 * HID_ + kb);
        acc[0] = __builtin_amdgcn_wmma_f32_16x16x32_f16(false, A, false, B0, (short)0, acc[0], false, false);
        acc[1] = __builtin_amdgcn_wmma_f32_16x16x32_f16(false, A, false, B1, (short)0, acc[1], true,  false);
        acc[2] = __builtin_amdgcn_wmma_f32_16x16x32_f16(false, A, false, B2, (short)0, acc[2], true,  false);
        acc[3] = __builtin_amdgcn_wmma_f32_16x16x32_f16(false, A, false, B3, (short)0, acc[3], true,  false);
      }
    }
  }

  // C/D layout: VGPR v -> M = v + hi*8, N = m
  const int posBase = s * D_ + dq * 16;
#pragma unroll
  for (int j = 0; j < 4; ++j) {
    const int n = j * 16 + m;
    const float bias = b2[n];
#pragma unroll
    for (int v = 0; v < 8; ++v) {
      const int pos = posBase + v + hi * 8;
      float val = acc[j][v] + bias;
      val = val > 0.f ? val : 0.f;
      h2[((size_t)b * S * D_ + pos) * HID_ + n] = (_Float16)val;
    }
  }
}

// -------- projection with LDS-shared A tile ------------------------------------------------
// Block = 256 threads (8 waves). Block computes 32 rows x all 512 N columns.
// A slab (32 rows x 128 K-halfs) staged in LDS once per stage and consumed by all 8 waves
// -> h2 streamed from HBM exactly once. Rows padded (+8 halfs) to spread LDS banks.
#define KT_   128
#define APAD_ (KT_ + 8)

__global__ void proj_wmma(const _Float16* __restrict__ h2,
                          const _Float16* __restrict__ wpT,
                          const float* __restrict__ bp,
                          float* __restrict__ out, int rows) {
  __shared__ _Float16 atile[32][APAD_];

  const int tid  = threadIdx.x;
  const int lane = tid & 31;
  const int wave = tid >> 5;          // 0..7 -> N block of 64
  const int m  = lane & 15;
  const int hi = lane >> 4;
  const int mt = blockIdx.x;          // 32-row tile over B*S
  const int n0 = wave << 6;
  const int K  = D_ * HID_;           // 5120

  // cooperative A staging: thread -> (row = tid/8, 16-half segment = tid%8)
  const int lrow = tid >> 3;
  const int lseg = (tid & 7) * 16;
  const int grow = ((mt * 32 + lrow) < rows) ? (mt * 32 + lrow) : (rows - 1);
  const _Float16* gsrc = h2 + (size_t)grow * K + lseg;

  const _Float16* bb0 = wpT + (size_t)(n0 +  0 + m) * K;
  const _Float16* bb1 = wpT + (size_t)(n0 + 16 + m) * K;
  const _Float16* bb2 = wpT + (size_t)(n0 + 32 + m) * K;
  const _Float16* bb3 = wpT + (size_t)(n0 + 48 + m) * K;

  v8f zero8 = {0.f,0.f,0.f,0.f,0.f,0.f,0.f,0.f};
  v8f acc[2][4];
#pragma unroll
  for (int i = 0; i < 2; ++i)
#pragma unroll
    for (int j = 0; j < 4; ++j) acc[i][j] = zero8;

  // preload stage 0 into registers
  u32x4 st0 = *reinterpret_cast<const u32x4*>(gsrc);
  u32x4 st1 = *reinterpret_cast<const u32x4*>(gsrc + 8);

  const int nStages = K / KT_;        // 40
  for (int kt = 0; kt < nStages; ++kt) {
    __syncthreads();                  // previous stage's readers done
    *reinterpret_cast<u32x4*>(&atile[lrow][lseg])     = st0;
    *reinterpret_cast<u32x4*>(&atile[lrow][lseg + 8]) = st1;
    __syncthreads();                  // slab visible to all waves

    if (kt + 1 < nStages) {           // issue next stage's global loads early
      const _Float16* gnxt = gsrc + (kt + 1) * KT_;
      st0 = *reinterpret_cast<const u32x4*>(gnxt);
      st1 = *reinterpret_cast<const u32x4*>(gnxt + 8);
    }

#pragma unroll
    for (int ks = 0; ks < KT_ / 32; ++ks) {
      const int kk = ks * 32 + hi * 16;
      const int kb = kt * KT_ + ks * 32 + hi * 16;
      v16h A0 = lds_frag8x2(&atile[m][kk],      &atile[m][kk + 8]);
      v16h A1 = lds_frag8x2(&atile[m + 16][kk], &atile[m + 16][kk + 8]);
      v16h B0 = load_frag16(bb0 + kb);
      v16h B1 = load_frag16(bb1 + kb);
      v16h B2 = load_frag16(bb2 + kb);
      v16h B3 = load_frag16(bb3 + kb);
      acc[0][0] = __builtin_amdgcn_wmma_f32_16x16x32_f16(false, A0, false, B0, (short)0, acc[0][0], false, false);
      acc[0][1] = __builtin_amdgcn_wmma_f32_16x16x32_f16(false, A0, false, B1, (short)0, acc[0][1], true,  false);
      acc[0][2] = __builtin_amdgcn_wmma_f32_16x16x32_f16(false, A0, false, B2, (short)0, acc[0][2], true,  false);
      acc[0][3] = __builtin_amdgcn_wmma_f32_16x16x32_f16(false, A0, false, B3, (short)0, acc[0][3], true,  false);
      acc[1][0] = __builtin_amdgcn_wmma_f32_16x16x32_f16(false, A1, false, B0, (short)0, acc[1][0], false, false);
      acc[1][1] = __builtin_amdgcn_wmma_f32_16x16x32_f16(false, A1, false, B1, (short)0, acc[1][1], true,  false);
      acc[1][2] = __builtin_amdgcn_wmma_f32_16x16x32_f16(false, A1, false, B2, (short)0, acc[1][2], true,  false);
      acc[1][3] = __builtin_amdgcn_wmma_f32_16x16x32_f16(false, A1, false, B3, (short)0, acc[1][3], true,  false);
    }
  }

#pragma unroll
  for (int sub = 0; sub < 2; ++sub) {
    const int rbase = mt * 32 + sub * 16 + hi * 8;
#pragma unroll
    for (int j = 0; j < 4; ++j) {
      const int n = n0 + j * 16 + m;
      const float bias = bp[n];
#pragma unroll
      for (int v = 0; v < 8; ++v) {
        const int row = rbase + v;
        if (row < rows) out[(size_t)row * OUT_ + n] = acc[sub][j][v] + bias;
      }
    }
  }
}

extern "C" void kernel_launch(void* const* d_in, const int* in_sizes, int n_in,
                              void* d_out, int out_size, void* d_ws, size_t ws_size,
                              hipStream_t stream) {
  (void)in_sizes; (void)n_in;
  const float* x    = (const float*)d_in[0];
  const int*   xlen = (const int*)d_in[1];
  const int*   bnd  = (const int*)d_in[2];
  const float* w1   = (const float*)d_in[3];
  const float* b1   = (const float*)d_in[4];
  const float* w2   = (const float*)d_in[5];
  const float* b2   = (const float*)d_in[6];
  const float* wp   = (const float*)d_in[7];
  const float* bp   = (const float*)d_in[8];
  float* out = (float*)d_out;

  // Recover S (max segments) from the flat output size: B*S*OUT (+ B for out_len)
  int S, write_len;
  if (out_size % (B_ * OUT_) == 0) { S = out_size / (B_ * OUT_); write_len = 0; }
  else { S = (out_size - B_) / (B_ * OUT_); write_len = 1; }
  if (S <= 0) return;

  // workspace carve (256B aligned)
  char* ws = (char*)d_ws;
  size_t off = 0;
  auto carve = [&](size_t bytes) -> void* {
    void* p = ws + off;
    off = (off + bytes + 255) & ~(size_t)255;
    return p;
  };
  int*       seg_s  = (int*)carve(sizeof(int) * (size_t)B_ * S);
  int*       seg_e  = (int*)carve(sizeof(int) * (size_t)B_ * S);
  float*     pooled = (float*)carve(sizeof(float) * (size_t)B_ * S * D_);
  _Float16*  h1     = (_Float16*)carve(sizeof(_Float16) * (size_t)B_ * S * D_ * HID_);
  _Float16*  h2     = (_Float16*)carve(sizeof(_Float16) * (size_t)B_ * S * D_ * HID_);
  _Float16*  wpT    = (_Float16*)carve(sizeof(_Float16) * (size_t)OUT_ * HID_ * D_);
  _Float16*  w2p    = (_Float16*)carve(sizeof(_Float16) * (size_t)9 * HID_ * HID_);
  if (off > ws_size) return; // not enough scratch

  prep_wpT<<<(HID_ * D_ * OUT_ + 255) / 256, 256, 0, stream>>>(wp, wpT);
  prep_w2p<<<(9 * HID_ * HID_ + 255) / 256, 256, 0, stream>>>(w2, w2p);
  seg_build<<<1, 32, 0, stream>>>(xlen, bnd, seg_s, seg_e,
      write_len ? (int*)(out + (size_t)B_ * S * OUT_) : (int*)nullptr, S, write_len);

  const int totalP = B_ * S * D_;
  pool_kernel<<<(totalP + 255) / 256, 256, 0, stream>>>(x, seg_s, seg_e, pooled, S);
  conv1_kernel<<<(totalP + 255) / 256, 256, 0, stream>>>(pooled, w1, b1, h1, S);

  conv2_wmma<<<dim3(B_ * S * 5), 32, 0, stream>>>(h1, w2p, b2, h2, S);

  const int rows = B_ * S;
  proj_wmma<<<dim3((rows + 31) / 32), 256, 0, stream>>>(h2, wpT, bp, out, rows);
}
